// ProjectedLocalCrossAttention_76158360093290
// MI455X (gfx1250) — matrix operand order repaired and checked by reading, not compiled, not run on hardware
//
#include <hip/hip_runtime.h>
#include <hip/hip_bf16.h>
#include <math.h>

#define G_TOTAL   100000
#define V_TOK     8
#define DIMA      256
#define HIDDEN_D  512
#define NHEADS    8
#define DHEAD     32
#define TILE_G    16   // gaussians per workgroup

typedef __attribute__((ext_vector_type(16))) __bf16       v16bf;
typedef __attribute__((ext_vector_type(8)))  float        v8f;
typedef __attribute__((ext_vector_type(4)))  unsigned int u32x4;

union FragBF { v16bf v; u32x4 q[2]; };

__device__ inline unsigned short f2bf_bits(float x) {
  unsigned u = __float_as_uint(x);
  u += 0x7FFFu + ((u >> 16) & 1u);          // round-to-nearest-even
  return (unsigned short)(u >> 16);
}
__device__ inline float bf2f_bits(unsigned short s) {
  return __uint_as_float(((unsigned)s) << 16);
}

// ---- WMMA fragment loaders (CDNA5 16-bit layouts, CDNA5 ISA) ----
// A (16x32 bf16): lane<16 -> M=lane, K = half*8 + [0..7] (VGPR0-3) and
//                 K = 16 + half*8 + [0..7] (VGPR4-7). Both runs contiguous.
__device__ inline void loadA(const unsigned short* base, int row0, int ld,
                             int kbase, int lane, FragBF& f) {
  int half = (lane >> 4) & 1;
  int m    = lane & 15;
  const unsigned short* p = base + (row0 + m) * ld + kbase + half * 8;
  f.q[0] = *(const u32x4*)(p);
  f.q[1] = *(const u32x4*)(p + 16);
}
// B (32x16 bf16) from W stored row-major [N][K]: lane -> N = n0 + (lane&15),
// K = kbase + half*8 + [0..7] and kbase + 16 + half*8 + [0..7] (same as A).
__device__ inline void loadB(const unsigned short* W, int n0, int ld,
                             int kbase, int lane, FragBF& f) {
  int half = (lane >> 4) & 1;
  int n    = lane & 15;
  const unsigned short* p = W + (size_t)(n0 + n) * ld + kbase + half * 8;
  f.q[0] = *(const u32x4*)(p);
  f.q[1] = *(const u32x4*)(p + 16);
}
__device__ inline v8f wmma_bf16(const FragBF& a, const FragBF& b, v8f c) {
  return __builtin_amdgcn_wmma_f32_16x16x32_bf16(false, a.v, false, b.v,
                                                 (short)0, c, false, false);
}

// ---------------- prep kernel: fold MHA in-projections into Wq/Wk/Wv ----------
// Wf[o][i] = sum_a Wi_part[o][a] * Wouter[a][i];  bf[o] = Wi_part[o]·bouter + bi[o]
__global__ void fuse_weights(const float* __restrict__ Wq, const float* __restrict__ bq,
                             const float* __restrict__ Wk, const float* __restrict__ bk,
                             const float* __restrict__ Wv, const float* __restrict__ bvv,
                             const float* __restrict__ Wi, const float* __restrict__ bi,
                             unsigned short* Wfq, unsigned short* Wfk, unsigned short* Wfv,
                             float* bfq, float* bfk, float* bfv) {
  int idx = blockIdx.x * blockDim.x + threadIdx.x;
  if (idx >= 3 * 256 * 256) return;
  int mat = idx >> 16;
  int rem = idx & 0xFFFF;
  int o   = rem >> 8;
  int i   = rem & 255;
  const float* Wout = (mat == 0) ? Wq : (mat == 1) ? Wk : Wv;
  const float* bout = (mat == 0) ? bq : (mat == 1) ? bk : bvv;
  const float* Wip  = Wi + (size_t)mat * 256 * 256;
  float acc = 0.f;
  for (int a = 0; a < 256; ++a) acc += Wip[o * 256 + a] * Wout[a * 256 + i];
  unsigned short* dst = (mat == 0) ? Wfq : (mat == 1) ? Wfk : Wfv;
  dst[o * 256 + i] = f2bf_bits(acc);
  if (i == 0) {
    float b = bi[mat * 256 + o];
    for (int a = 0; a < 256; ++a) b += Wip[o * 256 + a] * bout[a];
    float* bdst = (mat == 0) ? bfq : (mat == 1) ? bfk : bfv;
    bdst[o] = b;
  }
}

__global__ void convert_weights(const float* __restrict__ Wo, const float* __restrict__ W1,
                                unsigned short* Wob, unsigned short* W1b) {
  int idx = blockIdx.x * blockDim.x + threadIdx.x;
  if (idx < 65536)        Wob[idx]         = f2bf_bits(Wo[idx]);
  else if (idx < 196608)  W1b[idx - 65536] = f2bf_bits(W1[idx - 65536]);
}

// ---------------- fused main kernel: 16 gaussians per workgroup ----------------
// LDS layout (bytes):
//   Ag   [16x256 bf16]      @ 0       (8192)
//   At   [128x256 bf16]     @ 8192    (65536)
//   Qs   [16x256 bf16]      @ 73728   (8192)
//   Ks   [128x256 bf16]     @ 81920   (65536)
//   Vs   [128x256 bf16]     @ 147456  (65536)
//   CTX  [16x256 bf16]      @ 212992  (8192)
//   ATT  [16x8x8 f32]       @ 221184  (4096)
//   ATTD [16x256 f32]       @ 225280  (16384)
//   XN   [16x256 bf16]      @ 241664  (8192)
//   HS   [16x512 bf16]      @ 249856  (16384)   -> total 266240
#define SMEM_BYTES 266240
extern __shared__ unsigned char smem_raw[];

__global__ __launch_bounds__(256, 1)
void fused_attn(const float* __restrict__ gf, const float* __restrict__ lt,
                const unsigned char* __restrict__ vis,
                const unsigned short* __restrict__ Wfq, const unsigned short* __restrict__ Wfk,
                const unsigned short* __restrict__ Wfv, const unsigned short* __restrict__ Wob,
                const unsigned short* __restrict__ W1b,
                const float* __restrict__ bfq, const float* __restrict__ bfk,
                const float* __restrict__ bfv, const float* __restrict__ bo,
                const float* __restrict__ lng, const float* __restrict__ lnb,
                const float* __restrict__ b1, const float* __restrict__ W2,
                const float* __restrict__ b2,
                float* __restrict__ out_delta, float* __restrict__ out_w) {
  const int tid  = threadIdx.x;
  const int lane = tid & 31;
  const int wave = tid >> 5;
  const int g0   = blockIdx.x * TILE_G;

  unsigned short* Ag   = (unsigned short*)(smem_raw + 0);
  unsigned short* At   = (unsigned short*)(smem_raw + 8192);
  unsigned short* Qs   = (unsigned short*)(smem_raw + 73728);
  unsigned short* Ks   = (unsigned short*)(smem_raw + 81920);
  unsigned short* Vs   = (unsigned short*)(smem_raw + 147456);
  unsigned short* CTX  = (unsigned short*)(smem_raw + 212992);
  float*          ATT  = (float*)(smem_raw + 221184);
  float*          ATTD = (float*)(smem_raw + 225280);
  unsigned short* XN   = (unsigned short*)(smem_raw + 241664);
  unsigned short* HS   = (unsigned short*)(smem_raw + 249856);

  const int half = (lane >> 4) & 1;
  const int nn   = lane & 15;

  // ---- Phase 1: stage inputs to LDS as bf16 ----
  {
    const float* src = gf + (size_t)g0 * DIMA;
    for (int i = tid; i < TILE_G * DIMA; i += 256) Ag[i] = f2bf_bits(src[i]);
    const float* ts = lt + (size_t)g0 * V_TOK * DIMA;
    for (int i = tid; i < TILE_G * V_TOK * DIMA; i += 256) At[i] = f2bf_bits(ts[i]);
  }
  __syncthreads();

  // ---- Phase 2: Q = Ag @ Wfq^T + bfq  (16x256). Wave w owns N-tiles 2w,2w+1;
  //      A fragment hoisted across both N-tiles (shared M-tile). ----
  {
    int nt0 = wave * 2;
    v8f acc0 = {}, acc1 = {};
    for (int ks = 0; ks < 8; ++ks) {
      FragBF a, b0, b1f;
      loadA(Ag, 0, DIMA, ks * 32, lane, a);
      loadB(Wfq, nt0 * 16, DIMA, ks * 32, lane, b0);
      loadB(Wfq, nt0 * 16 + 16, DIMA, ks * 32, lane, b1f);
      acc0 = wmma_bf16(a, b0, acc0);
      acc1 = wmma_bf16(a, b1f, acc1);
    }
    float bias0 = bfq[nt0 * 16 + nn];
    float bias1 = bfq[nt0 * 16 + 16 + nn];
    for (int r = 0; r < 8; ++r) {
      Qs[(half * 8 + r) * DIMA + nt0 * 16 + nn]      = f2bf_bits(acc0[r] + bias0);
      Qs[(half * 8 + r) * DIMA + nt0 * 16 + 16 + nn] = f2bf_bits(acc1[r] + bias1);
    }
  }

  // ---- Phase 3: K,V = At(128x256) @ {Wfk,Wfv}^T + bias.
  //      32 column-GEMMs (16 N-tiles x 2 matrices) over 8 waves -> 4 each.
  //      Register-block all 8 M-tiles: B fragment loaded ONCE per k-step and
  //      reused by 8 WMMAs (8x fewer global B loads than tile-at-a-time). ----
  for (int p = wave; p < 32; p += 8) {
    int mat = p >> 4;
    int ct  = p & 15;
    const unsigned short* W   = mat ? Wfv : Wfk;
    const float*          bb  = mat ? bfv : bfk;
    unsigned short*       dst = mat ? Vs : Ks;
    v8f acc[8];
    for (int rt = 0; rt < 8; ++rt) acc[rt] = v8f{};
    for (int ks = 0; ks < 8; ++ks) {
      FragBF b;
      loadB(W, ct * 16, DIMA, ks * 32, lane, b);
#pragma unroll
      for (int rt = 0; rt < 8; ++rt) {
        FragBF a;
        loadA(At, rt * 16, DIMA, ks * 32, lane, a);
        acc[rt] = wmma_bf16(a, b, acc[rt]);
      }
    }
    float bias = bb[ct * 16 + nn];
#pragma unroll
    for (int rt = 0; rt < 8; ++rt)
      for (int r = 0; r < 8; ++r)
        dst[(rt * 16 + half * 8 + r) * DIMA + ct * 16 + nn] =
            f2bf_bits(acc[rt][r] + bias);
  }
  __syncthreads();

  // ---- Phase 4: per-(gaussian, head) attention over 8 tokens ----
  if (tid < TILE_G * NHEADS) {
    int g = tid >> 3, h = tid & 7;
    const unsigned char* vm = vis + (size_t)(g0 + g) * V_TOK;
    float qv[DHEAD];
    for (int d = 0; d < DHEAD; ++d) qv[d] = bf2f_bits(Qs[g * DIMA + h * DHEAD + d]);
    float sc[V_TOK];
    float mx = -1e30f;
    const float scale = 0.17677669529663687f;  // 1/sqrt(32)
    for (int v = 0; v < V_TOK; ++v) {
      float s = 0.f;
      const unsigned short* kr = Ks + (g * V_TOK + v) * DIMA + h * DHEAD;
      for (int d = 0; d < DHEAD; ++d) s += qv[d] * bf2f_bits(kr[d]);
      s *= scale;
      if (!vm[v]) s = -1e9f;
      sc[v] = s;
      mx = fmaxf(mx, s);
    }
    float den = 0.f;
    for (int v = 0; v < V_TOK; ++v) { sc[v] = expf(sc[v] - mx); den += sc[v]; }
    float inv = 1.f / den;
    for (int v = 0; v < V_TOK; ++v) {
      float p = sc[v] * inv;
      ATT[(g * NHEADS + h) * V_TOK + v] = p;
      sc[v] = p;
    }
    for (int d = 0; d < DHEAD; ++d) {
      float c = 0.f;
      for (int v = 0; v < V_TOK; ++v)
        c += sc[v] * bf2f_bits(Vs[(g * V_TOK + v) * DIMA + h * DHEAD + d]);
      CTX[g * DIMA + h * DHEAD + d] = f2bf_bits(c);
    }
  }
  __syncthreads();

  // weights output: mean over heads, zeroed when no view visible
  if (tid < TILE_G * V_TOK) {
    int g = tid >> 3, v = tid & 7;
    const unsigned char* vm = vis + (size_t)(g0 + g) * V_TOK;
    bool valid = false;
    for (int j = 0; j < V_TOK; ++j) valid = valid || (vm[j] != 0);
    float s = 0.f;
    for (int h = 0; h < NHEADS; ++h) s += ATT[(g * NHEADS + h) * V_TOK + v];
    out_w[(size_t)(g0 + g) * V_TOK + v] = valid ? (s * 0.125f) : 0.f;
  }

  // ---- Phase 5: attended = CTX @ Wo^T + bo -> f32 LDS (A hoisted) ----
  {
    int nt0 = wave * 2;
    v8f acc0 = {}, acc1 = {};
    for (int ks = 0; ks < 8; ++ks) {
      FragBF a, b0, b1f;
      loadA(CTX, 0, DIMA, ks * 32, lane, a);
      loadB(Wob, nt0 * 16, DIMA, ks * 32, lane, b0);
      loadB(Wob, nt0 * 16 + 16, DIMA, ks * 32, lane, b1f);
      acc0 = wmma_bf16(a, b0, acc0);
      acc1 = wmma_bf16(a, b1f, acc1);
    }
    float bias0 = bo[nt0 * 16 + nn];
    float bias1 = bo[nt0 * 16 + 16 + nn];
    for (int r = 0; r < 8; ++r) {
      ATTD[(half * 8 + r) * DIMA + nt0 * 16 + nn]      = acc0[r] + bias0;
      ATTD[(half * 8 + r) * DIMA + nt0 * 16 + 16 + nn] = acc1[r] + bias1;
    }
  }
  __syncthreads();

  // ---- LayerNorm (eps=1e-5), one row per thread ----
  if (tid < TILE_G) {
    const float* x = ATTD + tid * DIMA;
    float mu = 0.f;
    for (int i = 0; i < DIMA; ++i) mu += x[i];
    mu *= (1.f / DIMA);
    float var = 0.f;
    for (int i = 0; i < DIMA; ++i) { float d = x[i] - mu; var += d * d; }
    var *= (1.f / DIMA);
    float rstd = rsqrtf(var + 1e-5f);
    for (int i = 0; i < DIMA; ++i)
      XN[tid * DIMA + i] = f2bf_bits((x[i] - mu) * rstd * lng[i] + lnb[i]);
  }
  __syncthreads();

  // ---- Phase 6: h = GELU(XN @ W1^T + b1)  (16x512, 4 N-tiles/wave, A hoisted) ----
  {
    int nt0 = wave * 4;
    v8f acc[4];
    for (int j = 0; j < 4; ++j) acc[j] = v8f{};
    for (int ks = 0; ks < 8; ++ks) {
      FragBF a;
      loadA(XN, 0, DIMA, ks * 32, lane, a);
#pragma unroll
      for (int j = 0; j < 4; ++j) {
        FragBF b;
        loadB(W1b, (nt0 + j) * 16, DIMA, ks * 32, lane, b);
        acc[j] = wmma_bf16(a, b, acc[j]);
      }
    }
#pragma unroll
    for (int j = 0; j < 4; ++j) {
      float bias = b1[(nt0 + j) * 16 + nn];
      for (int r = 0; r < 8; ++r) {
        float xg = acc[j][r] + bias;
        float gg = 0.5f * xg * (1.f + erff(xg * 0.70710678118654752f));
        HS[(half * 8 + r) * HIDDEN_D + (nt0 + j) * 16 + nn] = f2bf_bits(gg);
      }
    }
  }
  __syncthreads();

  // ---- Phase 7: delta = h @ W2^T + b2 (tiny), zeroed for invalid gaussians ----
  if (tid < TILE_G * 2) {
    int g = tid >> 1, o = tid & 1;
    const unsigned char* vm = vis + (size_t)(g0 + g) * V_TOK;
    bool valid = false;
    for (int j = 0; j < V_TOK; ++j) valid = valid || (vm[j] != 0);
    float acc = b2[o];
    const float* w2r = W2 + o * HIDDEN_D;
    for (int j = 0; j < HIDDEN_D; ++j) acc += bf2f_bits(HS[g * HIDDEN_D + j]) * w2r[j];
    out_delta[(size_t)(g0 + g) * 2 + o] = valid ? acc : 0.f;
  }
}

extern "C" void kernel_launch(void* const* d_in, const int* in_sizes, int n_in,
                              void* d_out, int out_size, void* d_ws, size_t ws_size,
                              hipStream_t stream) {
  (void)in_sizes; (void)n_in; (void)out_size; (void)ws_size;
  const float* gf  = (const float*)d_in[0];
  const float* lt  = (const float*)d_in[1];
  const unsigned char* vis = (const unsigned char*)d_in[2];
  const float* Wq  = (const float*)d_in[3];
  const float* bq  = (const float*)d_in[4];
  const float* Wk  = (const float*)d_in[5];
  const float* bk  = (const float*)d_in[6];
  const float* Wv  = (const float*)d_in[7];
  const float* bv  = (const float*)d_in[8];
  const float* Wi  = (const float*)d_in[9];
  const float* bi  = (const float*)d_in[10];
  const float* Wo  = (const float*)d_in[11];
  const float* bo  = (const float*)d_in[12];
  const float* lng = (const float*)d_in[13];
  const float* lnb = (const float*)d_in[14];
  const float* W1  = (const float*)d_in[15];
  const float* b1  = (const float*)d_in[16];
  const float* W2  = (const float*)d_in[17];
  const float* b2  = (const float*)d_in[18];

  unsigned char* ws = (unsigned char*)d_ws;
  unsigned short* Wfq = (unsigned short*)(ws + 0);
  unsigned short* Wfk = (unsigned short*)(ws + 131072);
  unsigned short* Wfv = (unsigned short*)(ws + 262144);
  unsigned short* Wob = (unsigned short*)(ws + 393216);
  unsigned short* W1b = (unsigned short*)(ws + 524288);
  float* bfq = (float*)(ws + 786432);
  float* bfk = (float*)(ws + 787456);
  float* bfv = (float*)(ws + 788480);

  fuse_weights<<<768, 256, 0, stream>>>(Wq, bq, Wk, bk, Wv, bv, Wi, bi,
                                        Wfq, Wfk, Wfv, bfq, bfk, bfv);
  convert_weights<<<768, 256, 0, stream>>>(Wo, W1, Wob, W1b);

  float* out_delta = (float*)d_out;
  float* out_w     = out_delta + (size_t)G_TOTAL * 2;
  fused_attn<<<G_TOTAL / TILE_G, 256, SMEM_BYTES, stream>>>(
      gf, lt, vis, Wfq, Wfk, Wfv, Wob, W1b, bfq, bfk, bfv,
      bo, lng, lnb, b1, W2, b2, out_delta, out_w);
}
